// Decoder_51539607552479
// MI455X (gfx1250) — compile-verified, matrix operand order checked by reading, not run on hardware
//
#include <hip/hip_runtime.h>

typedef __attribute__((ext_vector_type(8)))  _Float16 v8h;
typedef __attribute__((ext_vector_type(16))) _Float16 v16h;
typedef __attribute__((ext_vector_type(8)))  float    v8f;
typedef __attribute__((ext_vector_type(4)))  int      v4i;

#define BN_EPS 1e-5f

// Halo'd activation layout: [b][66][66][512] f16, 1-pixel zero ring -> no
// boundary branches in the GEMM A-stage.
#define HW_H 66
#define F_HALVES ((size_t)8 * HW_H * HW_H * 512)
#define F_BYTES  (F_HALVES * 2)

// -------- async global->LDS copy (CDNA5) with sync fallback ----------------
#if defined(__gfx1250__) && __has_builtin(__builtin_amdgcn_global_load_async_to_lds_b128)
#define HAS_ASYNC_LDS 1
#else
#define HAS_ASYNC_LDS 0
#endif

__device__ __forceinline__ void cp_b128(const _Float16* g, _Float16* l) {
#if HAS_ASYNC_LDS
  typedef __attribute__((address_space(1))) v4i* g4p;  // global int4*
  typedef __attribute__((address_space(3))) v4i* l4p;  // LDS int4*
  __builtin_amdgcn_global_load_async_to_lds_b128(
      (g4p)(unsigned long long)(const void*)g,
      (l4p)(unsigned int)(unsigned long long)(void*)l, 0, 0);
#else
  *(v8h*)l = *(const v8h*)g;
#endif
}
// Allow the 8 just-issued next-stage copies to remain outstanding; retire the
// previous stage's 8.
__device__ __forceinline__ void wait_async_prev() {
#if HAS_ASYNC_LDS
#if __has_builtin(__builtin_amdgcn_s_wait_asynccnt)
  __builtin_amdgcn_s_wait_asynccnt(8);
#else
  asm volatile("s_wait_asynccnt 0x8" ::: "memory");
#endif
#endif
}
__device__ __forceinline__ void wait_async_all() {
#if HAS_ASYNC_LDS
#if __has_builtin(__builtin_amdgcn_s_wait_asynccnt)
  __builtin_amdgcn_s_wait_asynccnt(0);
#else
  asm volatile("s_wait_asynccnt 0x0" ::: "memory");
#endif
#endif
}

// ---------------------------------------------------------------------------
// Zero-fill (halo init for F0h/F1h).
// ---------------------------------------------------------------------------
__global__ __launch_bounds__(256) void zero_kernel(ulonglong2* __restrict__ p,
                                                   size_t n) {
  const ulonglong2 z = {0ull, 0ull};
  for (size_t i = (size_t)blockIdx.x * blockDim.x + threadIdx.x; i < n;
       i += (size_t)gridDim.x * blockDim.x)
    p[i] = z;
}

// ---------------------------------------------------------------------------
// Align-corners bilinear upsample of xt (8,512,32,32)->(64,64), fuse with
// alpha*xm, emit f16 NHWC-with-halo F0h[b][y+1][x+1][c].
// ---------------------------------------------------------------------------
__global__ __launch_bounds__(256) void fuse_upsample_kernel(
    const float* __restrict__ xt, const float* __restrict__ xm,
    const float* __restrict__ alpha, _Float16* __restrict__ F0) {
  const int by = blockIdx.x;
  const int b  = by >> 6;
  const int y  = by & 63;
  const float a = alpha[0];

  const float ry = 31.0f / 63.0f;
  const float fy = (float)y * ry;
  const int   y0 = (int)floorf(fy);
  const int   y1 = min(y0 + 1, 31);
  const float wy = fy - (float)y0;

  __shared__ _Float16 tile[64 * 66];
  const int t = threadIdx.x;

  for (int cc = 0; cc < 512; cc += 64) {
    #pragma unroll
    for (int i = 0; i < 16; ++i) {
      const int e  = t + i * 256;
      const int cl = e >> 6;
      const int x  = e & 63;
      const int c  = cc + cl;
      const float fx = (float)x * ry;
      const int   x0 = (int)floorf(fx);
      const int   x1 = min(x0 + 1, 31);
      const float wx = fx - (float)x0;
      const float* xtc = xt + ((size_t)b * 512 + c) * 1024;
      const float p00 = xtc[y0 * 32 + x0];
      const float p01 = xtc[y0 * 32 + x1];
      const float p10 = xtc[y1 * 32 + x0];
      const float p11 = xtc[y1 * 32 + x1];
      const float up  = (p00 * (1.f - wx) + p01 * wx) * (1.f - wy) +
                        (p10 * (1.f - wx) + p11 * wx) * wy;
      const float xmv = xm[(((size_t)b * 512 + c) * 64 + y) * 64 + x];
      tile[cl * 66 + x] = (_Float16)(up + a * xmv);
    }
    __syncthreads();
    #pragma unroll
    for (int i = 0; i < 16; ++i) {
      const int e  = t + i * 256;
      const int x  = e >> 6;
      const int cl = e & 63;
      F0[(((size_t)b * HW_H + (y + 1)) * HW_H + (x + 1)) * 512 + cc + cl] =
          tile[cl * 66 + x];
    }
    __syncthreads();
  }
}

// ---------------------------------------------------------------------------
// Weight repack: (Cout,512,3,3) fp32 -> f16 [kp][co][ci] (K contiguous).
// ---------------------------------------------------------------------------
__global__ __launch_bounds__(256) void wtrans_kernel(
    const float* __restrict__ w, _Float16* __restrict__ Wt, int cout) {
  const size_t total = (size_t)cout * 512 * 9;
  for (size_t o = (size_t)blockIdx.x * blockDim.x + threadIdx.x; o < total;
       o += (size_t)gridDim.x * blockDim.x) {
    const int ci  = (int)(o & 511);
    const size_t rem = o >> 9;
    const int co  = (int)(rem % cout);
    const int kp  = (int)(rem / cout);
    Wt[o] = (_Float16)w[((size_t)co * 512 + ci) * 9 + kp];
  }
}

// ---------------------------------------------------------------------------
// Implicit-GEMM 3x3 conv + BN + ReLU.
//   A: f16 halo NHWC (8,66,66,512) -> M = 32768, K = 9*512 = 4608
//   W: f16 [9][COUT][512]
// WG tile 128(M) x 128(N); 8 wave32s as 4x2; wave tile 32x64 = 2x4 WMMA
// (v_wmma_f32_16x16x32_f16). K-tile 64 (two 32-K WMMA steps per barrier
// pair), double-buffered LDS (73.7KB of the 320KB/WGP) filled by async
// global->LDS DMA, overlapped with compute via s_wait_asynccnt.
// OUTMODE 0: f16 halo NHWC (feeds next conv). OUTMODE 1: fp32 NCHW.
// ---------------------------------------------------------------------------
template <int COUT, int OUTMODE>
__global__ __launch_bounds__(256) void conv_gemm_kernel(
    const _Float16* __restrict__ A, const _Float16* __restrict__ W,
    const float* __restrict__ gm, const float* __restrict__ bt,
    const float* __restrict__ mn, const float* __restrict__ vr,
    _Float16* __restrict__ outH, float* __restrict__ outF) {
  __shared__ _Float16 sA[2][128 * 72];   // 128 M-rows x 64 K (pad 72)
  __shared__ _Float16 sB[2][128 * 72];   // 128 N-cols x 64 K (pad 72)

  const int t    = threadIdx.x;
  const int lane = t & 31;
  const int wave = t >> 5;
  const int wm   = wave & 3;             // M wave 0..3  (32 rows each)
  const int wn   = wave >> 2;            // N wave 0..1  (64 cols each)
  const int m0   = blockIdx.x * 128;
  const int n0   = blockIdx.y * 128;

  // A staging: thread -> (row r, 32-half chunk); base address is kt-invariant.
  const int ar = t >> 1;
  const int ao = (t & 1) * 32;
  const int am = m0 + ar;
  const int ab = am >> 12;
  const int ay = (am >> 6) & 63;
  const int ax = am & 63;
  const size_t abase =
      (((size_t)ab * HW_H + ay) * HW_H + ax) * 512 + ao;   // +1 halo folded by ky/kx
  // B staging: thread -> (col n, 32-half chunk)
  const int bn = t >> 1;
  const int bo = (t & 1) * 32;
  const size_t bbase = ((size_t)(n0 + bn)) * 512 + bo;

  auto issueStage = [&](int kt, int buf) {
    const int kp  = kt >> 3;             // ky*3+kx  (8 x 64-K tiles per plane)
    const int ci0 = (kt & 7) * 64;
    const int ky  = kp / 3;
    const int kx  = kp - ky * 3;
    const _Float16* asrc = A + abase + ((size_t)(ky * HW_H + kx)) * 512 + ci0;
    _Float16* adst = &sA[buf][ar * 72 + ao];
    cp_b128(asrc,      adst);
    cp_b128(asrc + 8,  adst + 8);
    cp_b128(asrc + 16, adst + 16);
    cp_b128(asrc + 24, adst + 24);
    const _Float16* wsrc = W + bbase + (size_t)kp * COUT * 512 + ci0;
    _Float16* bdst = &sB[buf][bn * 72 + bo];
    cp_b128(wsrc,      bdst);
    cp_b128(wsrc + 8,  bdst + 8);
    cp_b128(wsrc + 16, bdst + 16);
    cp_b128(wsrc + 24, bdst + 24);
  };

  const v8f zf = {0.f, 0.f, 0.f, 0.f, 0.f, 0.f, 0.f, 0.f};
  v8f acc[2][4];
  #pragma unroll
  for (int mi = 0; mi < 2; ++mi)
    #pragma unroll
    for (int ni = 0; ni < 4; ++ni) acc[mi][ni] = zf;

  const int lrow = lane & 15;
  const int base = (lane >> 4) * 8;      // K-half select per ISA 16-bit layout

  auto computeStage = [&](int buf) {
    #pragma unroll
    for (int ks = 0; ks < 64; ks += 32) {
      v16h af[2], bf[4];
      #pragma unroll
      for (int mi = 0; mi < 2; ++mi) {
        const int row = wm * 32 + mi * 16 + lrow;
        v8h lo = *(const v8h*)&sA[buf][row * 72 + ks + base];
        v8h hi = *(const v8h*)&sA[buf][row * 72 + ks + base + 16];
        af[mi] = __builtin_shufflevector(lo, hi, 0, 1, 2, 3, 4, 5, 6, 7, 8, 9,
                                         10, 11, 12, 13, 14, 15);
      }
      #pragma unroll
      for (int ni = 0; ni < 4; ++ni) {
        const int col = wn * 64 + ni * 16 + lrow;
        v8h lo = *(const v8h*)&sB[buf][col * 72 + ks + base];
        v8h hi = *(const v8h*)&sB[buf][col * 72 + ks + base + 16];
        bf[ni] = __builtin_shufflevector(lo, hi, 0, 1, 2, 3, 4, 5, 6, 7, 8, 9,
                                         10, 11, 12, 13, 14, 15);
      }
      #pragma unroll
      for (int mi = 0; mi < 2; ++mi)
        #pragma unroll
        for (int ni = 0; ni < 4; ++ni)
          acc[mi][ni] = __builtin_amdgcn_wmma_f32_16x16x32_f16(
              false, af[mi], false, bf[ni], (short)0, acc[mi][ni], false,
              false);
    }
  };

  constexpr int NT = (9 * 512) / 64;     // 72 K-tiles
  issueStage(0, 0);
  for (int kt = 0; kt < NT; ++kt) {
    const int cur = kt & 1;
    if (kt + 1 < NT) {
      issueStage(kt + 1, 1 - cur);       // DMA next stage, overlaps compute
      wait_async_prev();                 // retire stage kt's copies
    } else {
      wait_async_all();
    }
    __syncthreads();                     // stage kt visible to all waves
    computeStage(cur);
    if (kt + 1 < NT) __syncthreads();    // done reading buf before overwrite
  }

  // Epilogue: BN + ReLU.
  // C/D layout: lanes 0-15 -> N=lane, M=vgpr; lanes 16-31 -> N=lane-16, M=vgpr+8
  const int mbase = (lane >> 4) << 3;
  #pragma unroll
  for (int ni = 0; ni < 4; ++ni) {
    const int co = n0 + wn * 64 + ni * 16 + lrow;
    const float sc = gm[co] * rsqrtf(vr[co] + BN_EPS);
    const float sh = bt[co] - mn[co] * sc;
    #pragma unroll
    for (int mi = 0; mi < 2; ++mi) {
      #pragma unroll
      for (int r = 0; r < 8; ++r) {
        float val = fmaxf(acc[mi][ni][r] * sc + sh, 0.0f);
        const int mg = m0 + wm * 32 + mi * 16 + mbase + r;
        const int bi = mg >> 12, yy = (mg >> 6) & 63, xx = mg & 63;
        if (OUTMODE == 0) {
          outH[(((size_t)bi * HW_H + (yy + 1)) * HW_H + (xx + 1)) * 512 + co] =
              (_Float16)val;
        } else {
          outF[(((size_t)bi * COUT + co) << 12) + (yy << 6) + xx] = val;
        }
      }
    }
  }
}

// ---------------------------------------------------------------------------
extern "C" void kernel_launch(void* const* d_in, const int* in_sizes, int n_in,
                              void* d_out, int out_size, void* d_ws,
                              size_t ws_size, hipStream_t stream) {
  const float* xt    = (const float*)d_in[0];
  const float* xm    = (const float*)d_in[1];
  // d_in[2] = xl (unused by the reference output)
  const float* alpha = (const float*)d_in[3];
  const float* w0    = (const float*)d_in[4];
  const float* g0    = (const float*)d_in[5];
  const float* b0    = (const float*)d_in[6];
  const float* m0    = (const float*)d_in[7];
  const float* v0    = (const float*)d_in[8];
  const float* w1    = (const float*)d_in[9];
  const float* g1    = (const float*)d_in[10];
  const float* b1    = (const float*)d_in[11];
  const float* m1    = (const float*)d_in[12];
  const float* v1    = (const float*)d_in[13];

  const size_t WT0_BYTES = (size_t)9 * 512 * 512 * sizeof(_Float16);
  const size_t WT1_BYTES = (size_t)9 * 256 * 512 * sizeof(_Float16);
  if (ws_size < 2 * F_BYTES + WT0_BYTES + WT1_BYTES) return;

  char* ws = (char*)d_ws;
  _Float16* F0h = (_Float16*)(ws);
  _Float16* F1h = (_Float16*)(ws + F_BYTES);
  _Float16* Wt0 = (_Float16*)(ws + 2 * F_BYTES);
  _Float16* Wt1 = (_Float16*)(ws + 2 * F_BYTES + WT0_BYTES);

  // Zero both halo activation buffers (covers the 1-px ring; interiors are
  // overwritten below). 2 * 35.7MB at 23.3 TB/s ~= 3us.
  zero_kernel<<<4096, 256, 0, stream>>>((ulonglong2*)ws, (2 * F_BYTES) / 16);

  fuse_upsample_kernel<<<8 * 64, 256, 0, stream>>>(xt, xm, alpha, F0h);
  wtrans_kernel<<<2048, 256, 0, stream>>>(w0, Wt0, 512);
  wtrans_kernel<<<1024, 256, 0, stream>>>(w1, Wt1, 256);

  conv_gemm_kernel<512, 0><<<dim3(256, 4), 256, 0, stream>>>(
      F0h, Wt0, g0, b0, m0, v0, F1h, nullptr);
  conv_gemm_kernel<256, 1><<<dim3(256, 2), 256, 0, stream>>>(
      F1h, Wt1, g1, b1, m1, v1, nullptr, (float*)d_out);
}
